// LlamaAttention_13176959664571
// MI455X (gfx1250) — compile-verified, matrix operand order checked by reading, not run on hardware
//
#include <hip/hip_runtime.h>
#include <hip/hip_bf16.h>
#include <math.h>

// ---------------------------------------------------------------------------
// LlamaAttention forward, gfx1250 (MI455X). bf16 WMMA core, fragment loads
// are 2x ds_load_b128 (k-minor layout per ISA 7.12.2). GEMM staging uses
// CDNA5 async-to-LDS copies (ASYNCcnt) with double-buffered LDS.
// S=2048, H=4096, NH=32, D=128.
// ---------------------------------------------------------------------------

typedef __attribute__((ext_vector_type(16))) __bf16        v16bf;
typedef __attribute__((ext_vector_type(8)))  float         v8f;
typedef __attribute__((ext_vector_type(8)))  unsigned int  v8u;

#define S_LEN   2048
#define H_DIM   4096
#define N_HEADS 32
#define HEAD_D  128

static __device__ __forceinline__ unsigned short f2bf(float f) {
  unsigned u = __float_as_uint(f);
  u += 0x7FFFu + ((u >> 16) & 1u);          // round-to-nearest-even
  return (unsigned short)(u >> 16);
}
static __device__ __forceinline__ float bf2f(unsigned short s) {
  return __uint_as_float(((unsigned)s) << 16);
}
static __device__ __forceinline__ v8f vzero8() {
  v8f z;
#pragma unroll
  for (int i = 0; i < 8; ++i) z[i] = 0.0f;
  return z;
}
static __device__ __forceinline__ v8f wmma_bf16(v16bf a, v16bf b, v8f c) {
  return __builtin_amdgcn_wmma_f32_16x16x32_bf16(false, a, false, b,
                                                 (short)0, c, false, false);
}

// Async copy 16B global -> LDS (GVS mode), tracked by ASYNCcnt.
// lds_off: LDS byte offset; g_off: byte offset from base (signed 32-bit).
static __device__ __forceinline__ void async_ld_b128(unsigned lds_off,
                                                     int g_off,
                                                     const void* base) {
  asm volatile("global_load_async_to_lds_b128 %0, %1, %2"
               :: "v"(lds_off), "v"(g_off), "s"(base)
               : "memory");
}

// 16-bit fragment from LDS tile stored [row][k] (k contiguous).
// Per ISA 7.12.2: lane = (khalf<<4) | row%16 ; elements 0..7 are k=kh*8+0..7,
// elements 8..15 are k=16+kh*8+0..7  ->  two contiguous 8-short runs.
// strideShorts must be a multiple of 8 (16B alignment for ds_load_b128).
static __device__ __forceinline__ v16bf frag16(const unsigned short* base,
                                               int strideShorts, int rowbase,
                                               int cbase, int lane) {
  const int row = rowbase + (lane & 15);
  const int o   = row * strideShorts + cbase + ((lane >> 4) << 3);
  uint4 lo = *(const uint4*)(base + o);
  uint4 hi = *(const uint4*)(base + o + 16);
  v8u w;
  w[0] = lo.x; w[1] = lo.y; w[2] = lo.z; w[3] = lo.w;
  w[4] = hi.x; w[5] = hi.y; w[6] = hi.z; w[7] = hi.w;
  return __builtin_bit_cast(v16bf, w);
}

// ---------------------------------------------------------------------------
// fp32 -> bf16 straight conversion (X). One float4 per thread.
// ---------------------------------------------------------------------------
__global__ __launch_bounds__(256) void conv_f32_bf16(
    const float* __restrict__ in, unsigned short* __restrict__ out) {
  int i = blockIdx.x * 256 + threadIdx.x;
  float4 f = ((const float4*)in)[i];
  uint2 w;
  w.x = (unsigned)f2bf(f.x) | ((unsigned)f2bf(f.y) << 16);
  w.y = (unsigned)f2bf(f.z) | ((unsigned)f2bf(f.w) << 16);
  ((uint2*)out)[i] = w;
}

// ---------------------------------------------------------------------------
// fp32 [R][C] -> bf16 [C][R] (weight transpose, 64x64 LDS tiles).
// ---------------------------------------------------------------------------
__global__ __launch_bounds__(256) void transpose_f32_bf16(
    const float* __restrict__ in, unsigned short* __restrict__ out,
    int R, int C) {
  __shared__ unsigned short t[64 * 72];
  const int r0 = blockIdx.y * 64, c0 = blockIdx.x * 64;
#pragma unroll
  for (int p = 0; p < 4; ++p) {
    int i = p * 256 + threadIdx.x;          // 1024 float4 tasks
    int r = i >> 4, c4 = (i & 15) * 4;
    float4 f = *(const float4*)&in[(r0 + r) * C + c0 + c4];
    uint2 w;
    w.x = (unsigned)f2bf(f.x) | ((unsigned)f2bf(f.y) << 16);
    w.y = (unsigned)f2bf(f.z) | ((unsigned)f2bf(f.w) << 16);
    *(uint2*)&t[r * 72 + c4] = w;
  }
  __syncthreads();
#pragma unroll
  for (int p = 0; p < 2; ++p) {
    int i = p * 256 + threadIdx.x;          // 512 uint4 tasks
    int c = i >> 3, r8 = (i & 7) * 8;
    unsigned v[4];
#pragma unroll
    for (int j = 0; j < 4; ++j) {
      unsigned a = t[(r8 + 2 * j) * 72 + c];
      unsigned b = t[(r8 + 2 * j + 1) * 72 + c];
      v[j] = a | (b << 16);
    }
    *(uint4*)&out[(c0 + c) * R + r0 + r8] = make_uint4(v[0], v[1], v[2], v[3]);
  }
}

// ---------------------------------------------------------------------------
// bf16 [R][C] -> bf16 [C][R] (V transpose).
// ---------------------------------------------------------------------------
__global__ __launch_bounds__(256) void transpose_bf16(
    const unsigned short* __restrict__ in, unsigned short* __restrict__ out,
    int R, int C) {
  __shared__ unsigned short t[64 * 72];
  const int r0 = blockIdx.y * 64, c0 = blockIdx.x * 64;
#pragma unroll
  for (int p = 0; p < 2; ++p) {
    int i = p * 256 + threadIdx.x;          // 512 uint4 tasks
    int r = i >> 3, c8 = (i & 7) * 8;
    *(uint4*)&t[r * 72 + c8] = *(const uint4*)&in[(r0 + r) * C + c0 + c8];
  }
  __syncthreads();
#pragma unroll
  for (int p = 0; p < 2; ++p) {
    int i = p * 256 + threadIdx.x;
    int c = i >> 3, r8 = (i & 7) * 8;
    unsigned v[4];
#pragma unroll
    for (int j = 0; j < 4; ++j) {
      unsigned a = t[(r8 + 2 * j) * 72 + c];
      unsigned b = t[(r8 + 2 * j + 1) * 72 + c];
      v[j] = a | (b << 16);
    }
    *(uint4*)&out[(c0 + c) * R + r0 + r8] = make_uint4(v[0], v[1], v[2], v[3]);
  }
}

// ---------------------------------------------------------------------------
// GEMM: C[M,N] = A[M,K] @ Bt[N,K]^T. bf16 in, bf16 or f32 out.
// Block = 128 threads (4 waves), tile 128(M) x 64(N), K-step 32.
// Double-buffered LDS filled by global_load_async_to_lds_b128 (ASYNCcnt):
// 6 async b128 copies per thread per stage; stage k+1 overlaps WMMA on k.
// ---------------------------------------------------------------------------
static __device__ __forceinline__ void gemm_stage_async(
    const unsigned short* A, const unsigned short* Bt, unsigned ldsA,
    unsigned ldsB, int m0, int n0, int K, int kk, int tid) {
#pragma unroll
  for (int p = 0; p < 4; ++p) {             // A tile: 128x32 = 512 b128
    int i = p * 128 + tid;
    int r = i >> 2, c8 = (i & 3) * 8;
    async_ld_b128(ldsA + (unsigned)(r * 40 + c8) * 2,
                  ((m0 + r) * K + kk + c8) * 2, A);
  }
#pragma unroll
  for (int p = 0; p < 2; ++p) {             // B tile: 64x32 = 256 b128
    int i = p * 128 + tid;
    int r = i >> 2, c8 = (i & 3) * 8;
    async_ld_b128(ldsB + (unsigned)(r * 40 + c8) * 2,
                  ((n0 + r) * K + kk + c8) * 2, Bt);
  }
}

template <bool OUT_BF16>
__global__ __launch_bounds__(128) void gemm_bf16t(
    const unsigned short* __restrict__ A, const unsigned short* __restrict__ Bt,
    void* __restrict__ Cout, int M, int N, int K) {
  __shared__ unsigned short sAbuf[2][128 * 40];   // [m][k], pad 40
  __shared__ unsigned short sBbuf[2][64 * 40];    // [n][k], pad 40

  const int tid  = threadIdx.x;
  const int wave = tid >> 5;
  const int lane = tid & 31;
  const int m0   = blockIdx.y * 128;
  const int n0   = blockIdx.x * 64;

  unsigned ldsA[2], ldsB[2];
  ldsA[0] = (unsigned)(unsigned long long)&sAbuf[0][0];
  ldsA[1] = (unsigned)(unsigned long long)&sAbuf[1][0];
  ldsB[0] = (unsigned)(unsigned long long)&sBbuf[0][0];
  ldsB[1] = (unsigned)(unsigned long long)&sBbuf[1][0];

  v8f acc[2][4];
#pragma unroll
  for (int s = 0; s < 2; ++s)
#pragma unroll
    for (int j = 0; j < 4; ++j) acc[s][j] = vzero8();

  gemm_stage_async(A, Bt, ldsA[0], ldsB[0], m0, n0, K, 0, tid);

  const int KT = K >> 5;
  for (int kt = 0; kt < KT; ++kt) {
    const int cur = kt & 1, nxt = cur ^ 1;
    int knext = (kt + 1) << 5;
    if (knext >= K) knext = 0;              // wrapped dead stage on last iter
    gemm_stage_async(A, Bt, ldsA[nxt], ldsB[nxt], m0, n0, K, knext, tid);

    asm volatile("s_wait_asynccnt 0x6" ::: "memory");  // current stage landed
    __syncthreads();

    const unsigned short* cA = sAbuf[cur];
    const unsigned short* cB = sBbuf[cur];
    v16bf a0 = frag16(cA, 40, wave * 32, 0, lane);
    v16bf a1 = frag16(cA, 40, wave * 32 + 16, 0, lane);
#pragma unroll
    for (int j = 0; j < 4; ++j) {
      v16bf b = frag16(cB, 40, j * 16, 0, lane);
      acc[0][j] = wmma_bf16(a0, b, acc[0][j]);
      acc[1][j] = wmma_bf16(a1, b, acc[1][j]);
    }
    __syncthreads();   // all waves done reading before next stage overwrites
  }

  const int half = lane >> 4, ln = lane & 15;
#pragma unroll
  for (int s = 0; s < 2; ++s)
#pragma unroll
    for (int j = 0; j < 4; ++j)
#pragma unroll
      for (int r = 0; r < 8; ++r) {
        int row = m0 + wave * 32 + s * 16 + r + 8 * half;
        int col = n0 + j * 16 + ln;
        if (OUT_BF16)
          ((unsigned short*)Cout)[row * N + col] = f2bf(acc[s][j][r]);
        else
          ((float*)Cout)[row * N + col] = acc[s][j][r];
      }
}

// ---------------------------------------------------------------------------
// RoPE on bf16 Q,K. One thread per (s, head, j<64).
// ---------------------------------------------------------------------------
__global__ __launch_bounds__(256) void rope_bf16(
    const unsigned short* __restrict__ Q0, const unsigned short* __restrict__ K0,
    unsigned short* __restrict__ Qh, unsigned short* __restrict__ Kh) {
  int idx = blockIdx.x * 256 + threadIdx.x;
  int j = idx & 63;
  int h = (idx >> 6) & (N_HEADS - 1);
  int s = idx >> 11;
  float inv = __expf(-(float)j * 0.14391156731f);   // 10000^(-j/64)
  float ang = (float)s * inv;
  float sn, cs;
  __sincosf(ang, &sn, &cs);
  int base = s * H_DIM + h * HEAD_D;

  float q1 = bf2f(Q0[base + j]), q2 = bf2f(Q0[base + 64 + j]);
  Qh[base + j]      = f2bf(q1 * cs - q2 * sn);
  Qh[base + 64 + j] = f2bf(q2 * cs + q1 * sn);

  float k1 = bf2f(K0[base + j]), k2 = bf2f(K0[base + 64 + j]);
  Kh[base + j]      = f2bf(k1 * cs - k2 * sn);
  Kh[base + 64 + j] = f2bf(k2 * cs + k1 * sn);
}

// ---------------------------------------------------------------------------
// Flash-style causal attention, one wave per (head, 16-query block).
// Vt is pre-transposed [H][S] so PV B-fragments are k-minor too.
// ---------------------------------------------------------------------------
__global__ __launch_bounds__(32) void attn_fwd(
    const unsigned short* __restrict__ Qh, const unsigned short* __restrict__ Kh,
    const unsigned short* __restrict__ Vt, unsigned short* __restrict__ O) {
  const int h    = blockIdx.x;
  const int q0   = blockIdx.y * 16;
  const int lane = threadIdx.x;
  const int half = lane >> 4, ln = lane & 15;

  __shared__ unsigned short sQ[16 * 136];    // [m][c]
  __shared__ unsigned short sK[32 * 136];    // [key][c]
  __shared__ unsigned short sV[128 * 40];    // [d][key]
  __shared__ unsigned short sP[16 * 40];     // [m][key]

  for (int i = lane; i < 256; i += 32) {     // Q tile 16x128
    int r = i >> 4, c8 = (i & 15) * 8;
    *(uint4*)&sQ[r * 136 + c8] =
        *(const uint4*)&Qh[(q0 + r) * H_DIM + h * HEAD_D + c8];
  }

  v8f acc[8];
#pragma unroll
  for (int j = 0; j < 8; ++j) acc[j] = vzero8();
  float mstat[8], lstat[8];
#pragma unroll
  for (int r = 0; r < 8; ++r) { mstat[r] = -1e30f; lstat[r] = 0.0f; }

  const float scale = 0.08838834764831845f;  // 1/sqrt(128)
  const int kbmax = (q0 + 15) >> 5;

  for (int kb = 0; kb <= kbmax; ++kb) {
    const int kbase = kb * 32;
    for (int i = lane; i < 512; i += 32) {   // K tile 32x128
      int r = i >> 4, c8 = (i & 15) * 8;
      *(uint4*)&sK[r * 136 + c8] =
          *(const uint4*)&Kh[(kbase + r) * H_DIM + h * HEAD_D + c8];
    }
    for (int i = lane; i < 512; i += 32) {   // V tile 128(d) x 32(key)
      int d = i >> 2, c8 = (i & 3) * 8;
      *(uint4*)&sV[d * 40 + c8] =
          *(const uint4*)&Vt[(h * HEAD_D + d) * S_LEN + kbase + c8];
    }
    __syncthreads();

    v8f s0 = vzero8(), s1 = vzero8();
#pragma unroll
    for (int c0 = 0; c0 < 128; c0 += 32) {
      v16bf aq  = frag16(sQ, 136, 0,  c0, lane);
      v16bf bk0 = frag16(sK, 136, 0,  c0, lane);
      v16bf bk1 = frag16(sK, 136, 16, c0, lane);
      s0 = wmma_bf16(aq, bk0, s0);
      s1 = wmma_bf16(aq, bk1, s1);
    }

    float alpha[8];
#pragma unroll
    for (int r = 0; r < 8; ++r) {
      int qrow = q0 + r + 8 * half;
      float v0 = s0[r] * scale;
      float v1 = s1[r] * scale;
      if (kbase + ln > qrow)      v0 = -1e30f;
      if (kbase + 16 + ln > qrow) v1 = -1e30f;
      float mx = fmaxf(v0, v1);
#pragma unroll
      for (int off = 8; off >= 1; off >>= 1)
        mx = fmaxf(mx, __shfl_xor(mx, off, 32));
      float mnew = fmaxf(mstat[r], mx);
      alpha[r] = __expf(mstat[r] - mnew);
      float p0 = __expf(v0 - mnew);
      float p1 = __expf(v1 - mnew);
      float rs = p0 + p1;
#pragma unroll
      for (int off = 8; off >= 1; off >>= 1)
        rs += __shfl_xor(rs, off, 32);
      lstat[r] = lstat[r] * alpha[r] + rs;
      mstat[r] = mnew;
      sP[(r + 8 * half) * 40 + ln]      = f2bf(p0);
      sP[(r + 8 * half) * 40 + 16 + ln] = f2bf(p1);
    }
#pragma unroll
    for (int j = 0; j < 8; ++j)
#pragma unroll
      for (int r = 0; r < 8; ++r) acc[j][r] *= alpha[r];

    __syncthreads();

    v16bf ap = frag16(sP, 40, 0, 0, lane);
#pragma unroll
    for (int j = 0; j < 8; ++j) {
      v16bf bv = frag16(sV, 40, j * 16, 0, lane);   // rows = d slice
      acc[j] = wmma_bf16(ap, bv, acc[j]);
    }
    __syncthreads();
  }

#pragma unroll
  for (int j = 0; j < 8; ++j)
#pragma unroll
    for (int r = 0; r < 8; ++r)
      O[(q0 + r + 8 * half) * H_DIM + h * HEAD_D + j * 16 + ln] =
          f2bf(acc[j][r] / lstat[r]);
}

// ---------------------------------------------------------------------------
// Host-side orchestration
// ---------------------------------------------------------------------------
extern "C" void kernel_launch(void* const* d_in, const int* in_sizes, int n_in,
                              void* d_out, int out_size, void* d_ws,
                              size_t ws_size, hipStream_t stream) {
  (void)in_sizes; (void)n_in; (void)out_size; (void)ws_size;
  const float* X  = (const float*)d_in[0];
  const float* Wq = (const float*)d_in[3];
  const float* Wk = (const float*)d_in[4];
  const float* Wv = (const float*)d_in[5];
  const float* Wo = (const float*)d_in[6];

  char* ws = (char*)d_ws;
  const size_t MB16 = (size_t)S_LEN * H_DIM * 2;   // 16 MB (bf16 S x H)
  unsigned short* Xh = (unsigned short*)(ws);
  unsigned short* Wt = (unsigned short*)(ws + MB16);            // 32 MB
  unsigned short* Q0 = (unsigned short*)(ws + 3 * MB16);
  unsigned short* K0 = (unsigned short*)(ws + 4 * MB16);
  unsigned short* V0 = (unsigned short*)(ws + 5 * MB16);
  unsigned short* Qh = (unsigned short*)(ws + 6 * MB16);
  unsigned short* Kh = (unsigned short*)(ws + 7 * MB16);
  unsigned short* Vt = (unsigned short*)(ws + 8 * MB16);
  unsigned short* attnh = Q0;      // Q0 dead after rope_bf16

  dim3 tgrid(H_DIM / 64, H_DIM / 64);      // weight transpose tiles
  dim3 ggrid(H_DIM / 64, S_LEN / 128);     // GEMM blocks

  conv_f32_bf16<<<(S_LEN * H_DIM / 4) / 256, 256, 0, stream>>>(X, Xh);

  transpose_f32_bf16<<<tgrid, 256, 0, stream>>>(Wq, Wt, H_DIM, H_DIM);
  gemm_bf16t<true><<<ggrid, 128, 0, stream>>>(Xh, Wt, Q0, S_LEN, H_DIM, H_DIM);
  transpose_f32_bf16<<<tgrid, 256, 0, stream>>>(Wk, Wt, H_DIM, H_DIM);
  gemm_bf16t<true><<<ggrid, 128, 0, stream>>>(Xh, Wt, K0, S_LEN, H_DIM, H_DIM);
  transpose_f32_bf16<<<tgrid, 256, 0, stream>>>(Wv, Wt, H_DIM, H_DIM);
  gemm_bf16t<true><<<ggrid, 128, 0, stream>>>(Xh, Wt, V0, S_LEN, H_DIM, H_DIM);

  rope_bf16<<<(S_LEN * N_HEADS * 64) / 256, 256, 0, stream>>>(Q0, K0, Qh, Kh);
  transpose_bf16<<<dim3(H_DIM / 64, S_LEN / 64), 256, 0, stream>>>(
      V0, Vt, S_LEN, H_DIM);

  attn_fwd<<<dim3(N_HEADS, S_LEN / 16), 32, 0, stream>>>(Qh, Kh, Vt, attnh);

  transpose_f32_bf16<<<tgrid, 256, 0, stream>>>(Wo, Wt, H_DIM, H_DIM);
  gemm_bf16t<false><<<ggrid, 128, 0, stream>>>(attnh, Wt, (float*)d_out,
                                               S_LEN, H_DIM, H_DIM);
}
